// KDMLayer_81767587381522
// MI455X (gfx1250) — matrix operand order repaired and checked by reading, not compiled, not run on hardware
//
#include <hip/hip_runtime.h>
#include <hip/hip_bf16.h>

#define BS     256
#define N_IN   64
#define N_COMP 2048
#define DIM_X  1024
#define DIM_Y  128
#define KEPS   1e-12f

typedef __attribute__((ext_vector_type(16))) _Float16 v16h;
typedef __attribute__((ext_vector_type(2)))  _Float16 v2h;
typedef __attribute__((ext_vector_type(8)))  float    v8f;

union V16H {
    v16h v;
    _Float16 h[16];
    uint4 q[2];
};

// ---------------------------------------------------------------------------
// Kernel 0: sum of |c_w| -> ws[0]
// ---------------------------------------------------------------------------
__global__ void kdm_sum_abs_cw(const float* __restrict__ c_w, float* __restrict__ ws) {
    __shared__ float red[256];
    const int t = threadIdx.x;
    float s = 0.0f;
    for (int i = t; i < N_COMP; i += 256) s += fabsf(c_w[i]);
    red[t] = s;
    __syncthreads();
    for (int off = 128; off > 0; off >>= 1) {
        if (t < off) red[t] += red[t + off];
        __syncthreads();
    }
    if (t == 0) ws[0] = red[0];
}

// ---------------------------------------------------------------------------
// Kernel 1: per-component b2[j] = ||c_x[j]||^2 and cwn[j] = |c_w[j]| / sum.
// One wave (32 lanes) per component row. ws layout (floats):
//   ws[0] = sum|c_w| ; cwn = ws+16 .. +16+2047 ; b2 = ws+16+2048 ..
// ---------------------------------------------------------------------------
__global__ void kdm_precomp(const float* __restrict__ c_x,
                            const float* __restrict__ c_w,
                            float* __restrict__ ws) {
    const int wave = threadIdx.x >> 5;
    const int lane = threadIdx.x & 31;
    const int j = blockIdx.x * 8 + wave;           // gridDim.x = 256 -> j in [0,2048)
    const float* row = c_x + (size_t)j * DIM_X;
    float ss = 0.0f;
    for (int d = lane; d < DIM_X; d += 32) {
        float v = row[d];
        ss += v * v;
    }
    for (int off = 16; off > 0; off >>= 1) ss += __shfl_down(ss, off);
    if (lane == 0) {
        ws[16 + N_COMP + j] = ss;
        ws[16 + j] = fabsf(c_w[j]) / fmaxf(ws[0], KEPS);
    }
}

// ---------------------------------------------------------------------------
// Kernel 2: main WMMA kernel. One block (8 waves, 256 thr) per batch b.
// Two GEMM passes over j (denominators, then normalized weighted reduce).
// ---------------------------------------------------------------------------
__global__ void __launch_bounds__(256)
kdm_main(const float* __restrict__ rho_in, const float* __restrict__ c_x,
         const float* __restrict__ sigma_p, const float* __restrict__ ws,
         float* __restrict__ out) {
    __shared__ alignas(16) _Float16 As[N_IN][DIM_X + 8];   // f16 A, padded rows
    __shared__ float a2p[N_IN][4];
    __shared__ float dpart[8][N_IN];
    __shared__ float a2[N_IN], inw[N_IN], denom[N_IN], coef[N_IN];

    const int b = blockIdx.x;
    const int t = threadIdx.x;
    const float* cwn = ws + 16;
    const float* b2v = ws + 16 + N_COMP;

    if (t < N_IN) {
        inw[t] = rho_in[((size_t)b * N_IN + t) * (DIM_X + 1)];
    }

    // Stage A into LDS as f16; 4 threads per row, 256 cols each; partial a^2.
    {
        const int row = t >> 2, seg = t & 3;
        const float* src = rho_in + ((size_t)b * N_IN + row) * (DIM_X + 1) + 1 + seg * 256;
        float ss = 0.0f;
        for (int c = 0; c < 256; c += 2) {
            float v0 = src[c], v1 = src[c + 1];
            ss += v0 * v0 + v1 * v1;
            v2h p; p.x = (_Float16)v0; p.y = (_Float16)v1;
            *(v2h*)&As[row][seg * 256 + c] = p;
        }
        a2p[row][seg] = ss;
    }
    __syncthreads();
    if (t < N_IN) {
        a2[t] = (a2p[t][0] + a2p[t][1]) + (a2p[t][2] + a2p[t][3]);
    }
    __syncthreads();

    const float sig = sigma_p[0];
    const float inv2s2 = 1.0f / (2.0f * sig * sig);

    const int wave  = t >> 5;
    const int lane  = t & 31;
    const int nHalf = lane >> 4;     // 0: K 0..15 half, 1: K 16..31 half (B layout)
    const int nIdx  = lane & 15;     // column within 16-wide tile

    for (int pass = 0; pass < 2; ++pass) {
        float dsum[4][8];
#pragma unroll
        for (int mt = 0; mt < 4; ++mt)
#pragma unroll
            for (int r = 0; r < 8; ++r) dsum[mt][r] = 0.0f;

        for (int jt = wave; jt < N_COMP / 16; jt += 8) {
            const int j0 = jt * 16;

            v8f acc[4];
#pragma unroll
            for (int mt = 0; mt < 4; ++mt)
#pragma unroll
                for (int r = 0; r < 8; ++r) acc[mt][r] = 0.0f;

            const float* brow = c_x + (size_t)(j0 + nIdx) * DIM_X + nHalf * 16;
            for (int k0 = 0; k0 < DIM_X; k0 += 32) {
                // B fragment: lane = column n, 16 consecutive K values per lane.
                const float4* bp = (const float4*)(brow + k0);
                float4 f0 = bp[0], f1 = bp[1], f2 = bp[2], f3 = bp[3];
                V16H bf;
                bf.h[0] = (_Float16)f0.x; bf.h[1] = (_Float16)f0.y;
                bf.h[2] = (_Float16)f0.z; bf.h[3] = (_Float16)f0.w;
                bf.h[4] = (_Float16)f1.x; bf.h[5] = (_Float16)f1.y;
                bf.h[6] = (_Float16)f1.z; bf.h[7] = (_Float16)f1.w;
                bf.h[8]  = (_Float16)f2.x; bf.h[9]  = (_Float16)f2.y;
                bf.h[10] = (_Float16)f2.z; bf.h[11] = (_Float16)f2.w;
                bf.h[12] = (_Float16)f3.x; bf.h[13] = (_Float16)f3.y;
                bf.h[14] = (_Float16)f3.z; bf.h[15] = (_Float16)f3.w;

#pragma unroll
                for (int mt = 0; mt < 4; ++mt) {
                    const int m = mt * 16 + nIdx;   // A lane = row m
                    V16H af;                         // 16 consecutive K per lane (matches B)
                    af.q[0] = *(const uint4*)&As[m][k0 + nHalf * 16];
                    af.q[1] = *(const uint4*)&As[m][k0 + nHalf * 16 + 8];
                    acc[mt] = __builtin_amdgcn_wmma_f32_16x16x32_f16(
                        false, af.v, false, bf.v, (short)0, acc[mt], false, false);
                }
            }

            // Epilogue: D element (vgpr r, lane) = S[m][n], m = mt*16 + r + 8*nHalf,
            // n = j0 + nIdx.
            const float b2n = b2v[j0 + nIdx];
            const float cw  = cwn[j0 + nIdx];
            if (pass == 0) {
#pragma unroll
                for (int mt = 0; mt < 4; ++mt) {
#pragma unroll
                    for (int r = 0; r < 8; ++r) {
                        const int m = mt * 16 + r + 8 * nHalf;
                        float ab = acc[mt][r];
                        float d2 = fmaxf(a2[m] + b2n - 2.0f * ab, 0.0f);
                        float K  = __expf(-d2 * inv2s2);
                        float s  = fmaxf(cw * K * K, KEPS);
                        s += __shfl_xor(s, 1);
                        s += __shfl_xor(s, 2);
                        s += __shfl_xor(s, 4);
                        s += __shfl_xor(s, 8);   // sum over the 16 columns of this tile
                        dsum[mt][r] += s;
                    }
                }
            } else {
                float osum = 0.0f;
#pragma unroll
                for (int mt = 0; mt < 4; ++mt) {
#pragma unroll
                    for (int r = 0; r < 8; ++r) {
                        const int m = mt * 16 + r + 8 * nHalf;
                        float ab = acc[mt][r];
                        float d2 = fmaxf(a2[m] + b2n - 2.0f * ab, 0.0f);
                        float K  = __expf(-d2 * inv2s2);
                        float s  = fmaxf(cw * K * K, KEPS);
                        osum += coef[m] * s;
                    }
                }
                osum += __shfl_xor(osum, 16);    // combine the two m-halves
                if (nHalf == 0) {
                    out[((size_t)b * N_COMP + (j0 + nIdx)) * (DIM_Y + 1)] = osum;
                }
            }
        }

        if (pass == 0) {
            // Deterministic cross-wave reduction of denominators (no float atomics).
            if (nIdx == 0) {
#pragma unroll
                for (int mt = 0; mt < 4; ++mt)
#pragma unroll
                    for (int r = 0; r < 8; ++r)
                        dpart[wave][mt * 16 + r + 8 * nHalf] = dsum[mt][r];
            }
            __syncthreads();
            if (t < N_IN) {
                float d = 0.0f;
#pragma unroll
                for (int w = 0; w < 8; ++w) d += dpart[w][t];
                denom[t] = d;
                coef[t] = inw[t] / d;            // d >= N_COMP*KEPS > 0
            }
            __syncthreads();
        }
    }
}

// ---------------------------------------------------------------------------
// Kernel 3: broadcast c_y into out[:, :, 1:129]  (the 268 MB streaming part)
// ---------------------------------------------------------------------------
__global__ void kdm_broadcast_y(const float* __restrict__ c_y, float* __restrict__ out) {
    const size_t e = (size_t)blockIdx.x * 256 + threadIdx.x;  // < 256*2048*128
    const int    d  = (int)(e & 127);
    const size_t rj = e >> 7;              // b*2048 + j
    const int    j  = (int)(rj & (N_COMP - 1));
    out[rj * (DIM_Y + 1) + 1 + d] = c_y[(size_t)j * DIM_Y + d];
}

// ---------------------------------------------------------------------------
extern "C" void kernel_launch(void* const* d_in, const int* in_sizes, int n_in,
                              void* d_out, int out_size, void* d_ws, size_t ws_size,
                              hipStream_t stream) {
    (void)in_sizes; (void)n_in; (void)out_size; (void)ws_size;
    const float* rho_in = (const float*)d_in[0];
    const float* c_x    = (const float*)d_in[1];
    const float* c_y    = (const float*)d_in[2];
    const float* c_w    = (const float*)d_in[3];
    const float* sigma  = (const float*)d_in[4];
    float* out = (float*)d_out;
    float* ws  = (float*)d_ws;   // needs (16 + 2*2048) floats ≈ 16.5 KB

    kdm_sum_abs_cw<<<1, 256, 0, stream>>>(c_w, ws);
    kdm_precomp<<<N_COMP / 8, 256, 0, stream>>>(c_x, c_w, ws);
    kdm_main<<<BS, 256, 0, stream>>>(rho_in, c_x, sigma, ws, out);

    const int total_y = BS * N_COMP * DIM_Y;                 // 67,108,864
    kdm_broadcast_y<<<total_y / 256, 256, 0, stream>>>(c_y, out);
}
